// PointAttention_36498632081492
// MI455X (gfx1250) — compile-verified, matrix operand order checked by reading, not run on hardware
//
#include <hip/hip_runtime.h>
#include <math.h>

#define H 256
#define LDSPAD 264  // We row stride in bf16 elements (padded vs 256 to dodge bank conflicts)

typedef __attribute__((ext_vector_type(16))) __bf16 v16bf;
typedef __attribute__((ext_vector_type(8)))  __bf16 v8bf;
typedef __attribute__((ext_vector_type(8)))  float  v8f;

// ---------------------------------------------------------------------------
// Kernel 1: qb[b,k] = bq[k] + be[k] + sum_j query[b,j] * Wq[k,j]
// Tiny (67 MFLOP) -> plain fp32.
// ---------------------------------------------------------------------------
__global__ void __launch_bounds__(256)
qb_kernel(const float* __restrict__ query, const float* __restrict__ Wq,
          const float* __restrict__ bq, const float* __restrict__ be,
          float* __restrict__ qb) {
  const int b = blockIdx.x;
  const int k = threadIdx.x;
  const float* qr = query + (size_t)b * (2 * H);
  const float* wr = Wq + (size_t)k * (2 * H);
  float s = bq[k] + be[k];
#pragma unroll 8
  for (int j = 0; j < 2 * H; ++j) s += qr[j] * wr[j];
  qb[b * H + k] = s;
}

// Convert 16 fp32 (four float4 chunks in per-lane A/B order) to a v16bf frag.
__device__ __forceinline__ v16bf cvt16(float4 a0, float4 a1, float4 a2, float4 a3) {
  v16bf r;
  r[0]  = (__bf16)a0.x; r[1]  = (__bf16)a0.y; r[2]  = (__bf16)a0.z; r[3]  = (__bf16)a0.w;
  r[4]  = (__bf16)a1.x; r[5]  = (__bf16)a1.y; r[6]  = (__bf16)a1.z; r[7]  = (__bf16)a1.w;
  r[8]  = (__bf16)a2.x; r[9]  = (__bf16)a2.y; r[10] = (__bf16)a2.z; r[11] = (__bf16)a2.w;
  r[12] = (__bf16)a3.x; r[13] = (__bf16)a3.y; r[14] = (__bf16)a3.z; r[15] = (__bf16)a3.w;
  return r;
}

// ---------------------------------------------------------------------------
// Kernel 2: per 16-row tile of ref (M = B*G rows):
//   e = ref_tile @ We^T  via v_wmma_f32_16x16x32_bf16 (K=256, 8 steps held in regs)
//   score[m] = sum_n tanh(qb[b,n] + e[m,n]) * v[n]
// We (bf16) lives in LDS (256 x 264 halfs), shared by all 8 waves.
// Loop order: nt outer with a single v8f accumulator -> ~110 live VGPRs, no spills.
// ---------------------------------------------------------------------------
__global__ void __launch_bounds__(256)
attn_kernel(const float* __restrict__ ref, const float* __restrict__ We,
            const float* __restrict__ v, const float* __restrict__ qb,
            float* __restrict__ out, int nTiles, int G) {
  extern __shared__ char smem[];
  float*  sV  = (float*)smem;                 // 256 f32 = 1 KB
  __bf16* sWe = (__bf16*)(smem + 1024);       // 256 * 264 bf16 = 132 KB

  const int tid = threadIdx.x;
  if (tid < H) sV[tid] = v[tid];
  // Stage We as bf16 into padded LDS rows (coalesced global reads).
  for (int e = tid; e < H * H; e += blockDim.x) {
    int n = e >> 8, h = e & (H - 1);
    sWe[n * LDSPAD + h] = (__bf16)We[e];
  }
  __syncthreads();

  const int lane = tid & 31;
  const int m    = lane & 15;   // row (A) / col (B) within the 16-wide half
  const int hi   = lane >> 4;   // which K-interleave half this lane holds
  const int wavesPerBlock = blockDim.x >> 5;
  const int gwave  = blockIdx.x * wavesPerBlock + (tid >> 5);
  const int nWaves = gridDim.x * wavesPerBlock;

  // Per-lane v values for this lane's 16 columns (n = nt*16 + m), kept in regs.
  float vv[16];
#pragma unroll
  for (int nt = 0; nt < 16; ++nt) vv[nt] = sV[nt * 16 + m];

  for (int tile = gwave; tile < nTiles; tile += nWaves) {
    // Opaque zero: makes all LDS B addresses loop-variant so LICM cannot hoist
    // the 256 ds_load_b128s out of the tile loop (which previously spilled to scratch).
    unsigned opq;
    asm volatile("v_mov_b32 %0, 0" : "=v"(opq));
    const __bf16* sWeT = sWe + opq;

    // Build all 8 A-fragments (whole K=256) for this 16-row tile: 64 VGPRs.
    const float* rowPtr = ref + (size_t)(tile * 16 + m) * H;
    v16bf af[8];
#pragma unroll
    for (int ks = 0; ks < 8; ++ks) {
      const float* pa = rowPtr + ks * 32 + hi * 8;
      float4 a0 = *(const float4*)(pa + 0);
      float4 a1 = *(const float4*)(pa + 4);
      float4 a2 = *(const float4*)(pa + 16);
      float4 a3 = *(const float4*)(pa + 20);
      af[ks] = cvt16(a0, a1, a2, a3);
    }

    const int bidx = (tile * 16) / G;        // G % 16 == 0 -> uniform per tile
    const float* qrow = qb + bidx * H;
    float part[8] = {0.f, 0.f, 0.f, 0.f, 0.f, 0.f, 0.f, 0.f};

#pragma unroll
    for (int nt = 0; nt < 16; ++nt) {
      v8f acc = (v8f){0.f, 0.f, 0.f, 0.f, 0.f, 0.f, 0.f, 0.f};
      const __bf16* pbBase = sWeT + (size_t)(nt * 16 + m) * LDSPAD + hi * 8;
#pragma unroll
      for (int ks = 0; ks < 8; ++ks) {
        // B fragment: lane's column n = nt*16 + m; B(K=h, N=n) = We[n][h]
        v8bf blo = *(const v8bf*)(pbBase + ks * 32);
        v8bf bhi = *(const v8bf*)(pbBase + ks * 32 + 16);
        v16bf bf = __builtin_shufflevector(blo, bhi, 0, 1, 2, 3, 4, 5, 6, 7,
                                           8, 9, 10, 11, 12, 13, 14, 15);
        acc = __builtin_amdgcn_wmma_f32_16x16x32_bf16(
            false, af[ks], false, bf, (short)0, acc, false, false);
      }
      // Immediate epilogue for this nt: D layout — VGPR r, lane l -> (M=r+8*hi, N=l&15)
      const float qv = qrow[nt * 16 + m];
      const float vn = vv[nt];
#pragma unroll
      for (int r = 0; r < 8; ++r) part[r] += tanhf(qv + acc[r]) * vn;
    }

    // Reduce over the 16 lanes of each half (N dimension).
#pragma unroll
    for (int r = 0; r < 8; ++r) {
      float s = part[r];
      s += __shfl_xor(s, 1, 32);
      s += __shfl_xor(s, 2, 32);
      s += __shfl_xor(s, 4, 32);
      s += __shfl_xor(s, 8, 32);
      part[r] = s;
    }
    if (m == 0) {
      const int rowBase = tile * 16 + hi * 8;
#pragma unroll
      for (int r = 0; r < 8; ++r) out[rowBase + r] = part[r];
    }
  }
}

// ---------------------------------------------------------------------------
extern "C" void kernel_launch(void* const* d_in, const int* in_sizes, int n_in,
                              void* d_out, int out_size, void* d_ws, size_t ws_size,
                              hipStream_t stream) {
  const float* query = (const float*)d_in[0];
  const float* ref   = (const float*)d_in[1];
  const float* Wq    = (const float*)d_in[2];
  const float* bq    = (const float*)d_in[3];
  const float* We    = (const float*)d_in[4];
  const float* be    = (const float*)d_in[5];
  const float* v     = (const float*)d_in[6];
  float* out = (float*)d_out;

  const int B = in_sizes[0] / (2 * H);           // 256
  const int G = in_sizes[1] / (B * H);           // 2000
  float* qb = (float*)d_ws;                      // B*H fp32 scratch

  qb_kernel<<<B, 256, 0, stream>>>(query, Wq, bq, be, qb);

  const int nTiles = (B * G) / 16;               // 32000 (G % 16 == 0)
  const int blocks = 500;                        // 4000 waves -> 8 tiles/wave
  const size_t shmem = 1024 + (size_t)H * LDSPAD * sizeof(__bf16); // ~133 KB
  attn_kernel<<<blocks, 256, shmem, stream>>>(ref, We, v, qb, out, nTiles, G);
}
